// GCNLayer_72224170050097
// MI455X (gfx1250) — matrix operand order repaired and checked by reading, not compiled, not run on hardware
//
#include <hip/hip_runtime.h>

typedef __attribute__((ext_vector_type(16))) __bf16 v16bf;
typedef __attribute__((ext_vector_type(8)))  __bf16 v8bf;
typedef __attribute__((ext_vector_type(8)))  float  v8f;
typedef __attribute__((ext_vector_type(4)))  float  f4;
typedef __attribute__((ext_vector_type(4)))  unsigned int u4;

namespace {
constexpr int kN     = 8192;
constexpr int kD     = 128;
constexpr int kBM    = 128;            // rows per stage-1 block tile
constexpr int kBK    = 32;             // K per WMMA step
constexpr int kSplit = 4;              // split-K factor
constexpr int kKC    = kN / kSplit;    // 2048 K per block
constexpr int kSteps = kKC / kBK;      // 64 steps
constexpr int kAStr  = 40;             // padded LDS row stride (bf16) for 32-wide tiles (80 B)
constexpr int kTStr  = 136;            // padded LDS row stride (bf16) for 128-wide tiles (272 B)
}

__device__ __forceinline__ unsigned short f2bfbits(float f) {
    unsigned int u = __float_as_uint(f);
    u += 0x7FFFu + ((u >> 16) & 1u);          // round-to-nearest-even
    return (unsigned short)(u >> 16);
}
__device__ __forceinline__ unsigned int packbf2(float a, float b) {
    return (unsigned int)f2bfbits(a) | ((unsigned int)f2bfbits(b) << 16);
}
// Build a v16bf WMMA fragment from two contiguous 16-byte chunks (two b128 loads).
__device__ __forceinline__ v16bf combine16(const __bf16* lo, const __bf16* hi) {
    v8bf a = *(const v8bf*)lo;
    v8bf b = *(const v8bf*)hi;
    v16bf r;
#pragma unroll
    for (int i = 0; i < 8; ++i) { r[i] = a[i]; r[i + 8] = b[i]; }
    return r;
}

// ---------------------------------------------------------------------------
// Kernel 1: d[i] = rsqrt(1 + sum_j adj[i][j])      (256 MB NT stream)
// ---------------------------------------------------------------------------
__global__ __launch_bounds__(256) void k_rowsum(const float* __restrict__ adj,
                                                float* __restrict__ dinv) {
    __shared__ float red[256];
    const int row = blockIdx.x;
    const f4* p = (const f4*)(adj + (size_t)row * kN);
    float s = 0.f;
#pragma unroll
    for (int i = 0; i < 8; ++i) {
        f4 v = __builtin_nontemporal_load(&p[threadIdx.x + 256 * i]);
        s += v.x + v.y + v.z + v.w;
    }
    red[threadIdx.x] = s;
    __syncthreads();
#pragma unroll
    for (int off = 128; off > 0; off >>= 1) {
        if ((int)threadIdx.x < off) red[threadIdx.x] += red[threadIdx.x + off];
        __syncthreads();
    }
    if (threadIdx.x == 0) dinv[row] = rsqrtf(red[0] + 1.0f);
}

// ---------------------------------------------------------------------------
// Kernel 2: xsT[c][k] = bf16(d[k]*x[k][c])  (transposed, 2 MB -> L2-resident)
//           Wbf[o][c] = bf16(W[o][c])
// ---------------------------------------------------------------------------
__global__ __launch_bounds__(256) void k_prep(const float* __restrict__ x,
                                              const float* __restrict__ W,
                                              const float* __restrict__ dinv,
                                              unsigned short* __restrict__ xsT,
                                              unsigned short* __restrict__ Wbf) {
    const int idx = blockIdx.x * 256 + threadIdx.x;
    const int nx = kN * kD;
    if (idx < nx) {
        const int c = idx >> 13;          // / 8192
        const int k = idx & (kN - 1);
        xsT[idx] = f2bfbits(dinv[k] * x[(size_t)k * kD + c]);
    } else if (idx < nx + kD * kD) {
        const int wi = idx - nx;
        Wbf[wi] = f2bfbits(W[wi]);
    }
}

// ---------------------------------------------------------------------------
// Kernel 3: split-K partial GEMM  Tpart[p] += adj[:, Kp] @ xs[Kp, :]
// grid (64, 4), 256 threads = 8 waves in a 4x2 tile grid over a 128x128 tile.
// ---------------------------------------------------------------------------
__global__ __launch_bounds__(256) void k_agg(const float* __restrict__ adj,
                                             const unsigned short* __restrict__ xsT,
                                             float* __restrict__ Tpart) {
    __shared__ __align__(16) __bf16 smem[20480];   // 40 KB: As[2][128*40] | Bs[2][128*40]
    __bf16* As = smem;
    __bf16* Bs = smem + 2 * kBM * kAStr;

    const int tid  = threadIdx.x;
    const int lane = tid & 31;
    const int l15  = lane & 15;
    const int lh   = lane >> 4;
    const int w    = tid >> 5;
    const int wr   = w & 3;        // wave row-group: rows 32*wr .. +31
    const int wc   = w >> 2;       // wave col-group: cols 64*wc .. +63
    const int m0   = blockIdx.x * kBM;
    const int kb   = blockIdx.y * kKC;

    const int ldr = tid >> 1;      // 0..127: A row / B (feature) row
    const int seg = tid & 1;       // 16-element half of the 32-wide K slab

    const v8f zero = {0.f, 0.f, 0.f, 0.f, 0.f, 0.f, 0.f, 0.f};
    v8f acc[2][4];
#pragma unroll
    for (int i = 0; i < 2; ++i)
#pragma unroll
        for (int t = 0; t < 4; ++t) acc[i][t] = zero;

    f4 pa[4];
    u4 pb[2];
    const float*          arow = adj + (size_t)(m0 + ldr) * kN + kb + seg * 16;
    const unsigned short* brow = xsT + (size_t)ldr * kN + kb + seg * 16;

    auto gload = [&](int k0) {                       // global -> regs (prefetch)
        const f4* ap = (const f4*)(arow + k0);
#pragma unroll
        for (int i = 0; i < 4; ++i) pa[i] = __builtin_nontemporal_load(&ap[i]);
        const u4* bp = (const u4*)(brow + k0);
        pb[0] = bp[0]; pb[1] = bp[1];
    };
    auto lstore = [&](int buf) {                     // regs -> LDS (fp32->bf16 for A)
        u4* ad = (u4*)(As + buf * (kBM * kAStr) + ldr * kAStr + seg * 16);
        u4 v0, v1;
        v0.x = packbf2(pa[0].x, pa[0].y); v0.y = packbf2(pa[0].z, pa[0].w);
        v0.z = packbf2(pa[1].x, pa[1].y); v0.w = packbf2(pa[1].z, pa[1].w);
        v1.x = packbf2(pa[2].x, pa[2].y); v1.y = packbf2(pa[2].z, pa[2].w);
        v1.z = packbf2(pa[3].x, pa[3].y); v1.w = packbf2(pa[3].z, pa[3].w);
        ad[0] = v0; ad[1] = v1;
        u4* bd = (u4*)(Bs + buf * (kD * kAStr) + ldr * kAStr + seg * 16);
        bd[0] = pb[0]; bd[1] = pb[1];
    };

    gload(0);
#pragma unroll 1
    for (int s = 0; s < kSteps; ++s) {
        const int buf = s & 1;
        lstore(buf);
        __syncthreads();                              // one barrier/iter (double buffer)
        if (s + 1 < kSteps) gload((s + 1) * kBK);     // overlap global latency w/ WMMA

        const __bf16* ab = As + buf * (kBM * kAStr);
        const __bf16* bb = Bs + buf * (kD * kAStr);
        v16bf af[2], bfr[4];
#pragma unroll
        for (int i = 0; i < 2; ++i) {
            // A 16x32 bf16 frag: lane=M; elems 0-7 K=lh*8.., elems 8-15 K=16+lh*8..
            const __bf16* r = ab + (32 * wr + 16 * i + l15) * kAStr;
            af[i] = combine16(r + lh * 8, r + 16 + lh * 8);
        }
#pragma unroll
        for (int t = 0; t < 4; ++t) {
            // B 32x16 bf16 frag: lane=N; elems 0-7 K=lh*8.., elems 8-15 K=16+lh*8..
            const __bf16* r = bb + (64 * wc + 16 * t + l15) * kAStr;
            bfr[t] = combine16(r + lh * 8, r + 16 + lh * 8);
        }
#pragma unroll
        for (int i = 0; i < 2; ++i)
#pragma unroll
            for (int t = 0; t < 4; ++t)
                acc[i][t] = __builtin_amdgcn_wmma_f32_16x16x32_bf16(
                    false, af[i], false, bfr[t], (short)0, acc[i][t], false, false);
    }

    // store f32 partial (D layout: VGPR r -> M = r + 8*lh, lane -> N)
    float* tp = Tpart + (size_t)blockIdx.y * (kN * kD);
#pragma unroll
    for (int i = 0; i < 2; ++i)
#pragma unroll
        for (int t = 0; t < 4; ++t) {
            const int c = 64 * wc + 16 * t + l15;
#pragma unroll
            for (int r = 0; r < 8; ++r) {
                const int M = 32 * wr + 16 * i + 8 * lh + r;
                tp[(size_t)(m0 + M) * kD + c] = acc[i][t][r];
            }
        }
}

// ---------------------------------------------------------------------------
// Kernel 4: reduce split-K partials + identity + row scale, then x Wbf^T + b
// 64 blocks x 256 threads; second GEMM done with bf16 WMMA from LDS/L2.
// ---------------------------------------------------------------------------
__global__ __launch_bounds__(256) void k_out(const float* __restrict__ Tpart,
                                             const float* __restrict__ x,
                                             const unsigned short* __restrict__ Wbf_u,
                                             const float* __restrict__ dinv,
                                             const float* __restrict__ bias,
                                             float* __restrict__ out) {
    __shared__ __align__(16) __bf16 Ts[kBM * kTStr];   // 34816 B
    const __bf16* Wbf = (const __bf16*)Wbf_u;

    const int tid = threadIdx.x;
    const int m0  = blockIdx.x * kBM;

    {   // T[row][c] = d[row] * (sum_p Tpart[p][row][c] + d[row]*x[row][c]) -> bf16 Ts
        const int row = tid >> 1;
        const int seg = tid & 1;
        const int gr  = m0 + row;
        const float dr = dinv[gr];
#pragma unroll
        for (int blk = 0; blk < 4; ++blk) {
            const int c0 = seg * 64 + blk * 16;
            f4 s[4];
#pragma unroll
            for (int j = 0; j < 4; ++j)
                s[j] = *(const f4*)(Tpart + (size_t)gr * kD + c0 + 4 * j);
#pragma unroll
            for (int p = 1; p < kSplit; ++p) {
                const float* tp = Tpart + (size_t)p * (kN * kD) + (size_t)gr * kD + c0;
#pragma unroll
                for (int j = 0; j < 4; ++j) s[j] += *(const f4*)(tp + 4 * j);
            }
#pragma unroll
            for (int j = 0; j < 4; ++j) {
                f4 xv = *(const f4*)(x + (size_t)gr * kD + c0 + 4 * j);
                s[j] = (s[j] + dr * xv) * dr;
            }
            u4* dst = (u4*)(Ts + row * kTStr + c0);
            u4 o0, o1;
            o0.x = packbf2(s[0].x, s[0].y); o0.y = packbf2(s[0].z, s[0].w);
            o0.z = packbf2(s[1].x, s[1].y); o0.w = packbf2(s[1].z, s[1].w);
            o1.x = packbf2(s[2].x, s[2].y); o1.y = packbf2(s[2].z, s[2].w);
            o1.z = packbf2(s[3].x, s[3].y); o1.w = packbf2(s[3].z, s[3].w);
            dst[0] = o0; dst[1] = o1;
        }
    }
    __syncthreads();

    const int lane = tid & 31;
    const int l15  = lane & 15;
    const int lh   = lane >> 4;
    const int w    = tid >> 5;
    const int wr   = w & 3;
    const int wc   = w >> 2;

    const v8f zero = {0.f, 0.f, 0.f, 0.f, 0.f, 0.f, 0.f, 0.f};
    v8f acc[2][4];
#pragma unroll
    for (int i = 0; i < 2; ++i)
#pragma unroll
        for (int t = 0; t < 4; ++t) acc[i][t] = zero;

#pragma unroll
    for (int ks = 0; ks < 4; ++ks) {      // K = feature dim, 4 steps of 32
        v16bf af[2], bw[4];
#pragma unroll
        for (int i = 0; i < 2; ++i) {
            const __bf16* r = Ts + (32 * wr + 16 * i + l15) * kTStr + 32 * ks;
            af[i] = combine16(r + lh * 8, r + 16 + lh * 8);
        }
#pragma unroll
        for (int t = 0; t < 4; ++t) {     // B = W^T: B[k=c][n=o] = W[o][c] (row-major fits)
            const __bf16* r = Wbf + (64 * wc + 16 * t + l15) * kD + 32 * ks;
            bw[t] = combine16(r + lh * 8, r + 16 + lh * 8);
        }
#pragma unroll
        for (int i = 0; i < 2; ++i)
#pragma unroll
            for (int t = 0; t < 4; ++t)
                acc[i][t] = __builtin_amdgcn_wmma_f32_16x16x32_bf16(
                    false, af[i], false, bw[t], (short)0, acc[i][t], false, false);
    }

#pragma unroll
    for (int i = 0; i < 2; ++i)
#pragma unroll
        for (int t = 0; t < 4; ++t) {
            const int c = 64 * wc + 16 * t + l15;
            const float bv = bias[c];
#pragma unroll
            for (int r = 0; r < 8; ++r) {
                const int M = 32 * wr + 16 * i + 8 * lh + r;
                out[(size_t)(m0 + M) * kD + c] = acc[i][t][r] + bv;
            }
        }
}

// ---------------------------------------------------------------------------
extern "C" void kernel_launch(void* const* d_in, const int* in_sizes, int n_in,
                              void* d_out, int out_size, void* d_ws, size_t ws_size,
                              hipStream_t stream) {
    (void)in_sizes; (void)n_in; (void)out_size; (void)ws_size;
    const float* x   = (const float*)d_in[0];   // [8192,128]
    const float* adj = (const float*)d_in[1];   // [8192,8192]
    const float* W   = (const float*)d_in[2];   // [128,128]
    const float* b   = (const float*)d_in[3];   // [128]
    float* out = (float*)d_out;                 // [8192,128]

    char* ws = (char*)d_ws;
    float*          ddeg = (float*)ws;                                      // 32 KB
    unsigned short* xsT  = (unsigned short*)(ws + 32768);                   // 2 MB
    unsigned short* Wbf  = (unsigned short*)(ws + 32768 + kN * kD * 2);     // 32 KB
    float*          Tp   = (float*)(ws + 32768 + kN * kD * 2 + kD * kD * 2);// 16 MB

    k_rowsum<<<kN, 256, 0, stream>>>(adj, ddeg);
    const int total = kN * kD + kD * kD;
    k_prep<<<(total + 255) / 256, 256, 0, stream>>>(x, W, ddeg, xsT, Wbf);
    dim3 g1(kN / kBM, kSplit);
    k_agg<<<g1, 256, 0, stream>>>(adj, xsT, Tp);
    k_out<<<kN / kBM, 256, 0, stream>>>(Tp, x, Wbf, ddeg, b, out);
}
